// GroupedQueryAttention_73693048864853
// MI455X (gfx1250) — matrix operand order repaired and checked by reading, not compiled, not run on hardware
//
#include <hip/hip_runtime.h>
#include <hip/hip_bf16.h>

// ---------------------------------------------------------------------------
// GQA forward for MI455X (gfx1250), f16 WMMA path with f32 accumulation.
// Staging uses GLOBAL_LOAD_ASYNC_TO_LDS_B128 (ASYNCcnt) when the toolchain
// exposes it, with double-buffered LDS tiles in the GEMM.
// ---------------------------------------------------------------------------

typedef __attribute__((ext_vector_type(16))) _Float16 v16h;
typedef __attribute__((ext_vector_type(8)))  _Float16 v8h;
typedef __attribute__((ext_vector_type(4)))  _Float16 v4h;
typedef __attribute__((ext_vector_type(8)))  float    v8f;

// Builtin async-to-LDS expects int4 pointers in explicit address spaces.
typedef int v4i __attribute__((__vector_size__(4 * sizeof(int))));
typedef __attribute__((address_space(1))) v4i* glob_v4i_ptr;
typedef __attribute__((address_space(3))) v4i* lds_v4i_ptr;

#define SEQ    4096
#define HIDDEN 2048
#define QH     16
#define KVH    4
#define HD     128
#define KVDIM  (KVH * HD)   // 512

#if defined(__AMDGCN__) && __has_builtin(__builtin_amdgcn_global_load_async_to_lds_b128)
#define ASYNC_LDS 1
#else
#define ASYNC_LDS 0
#endif

// Stage 16 halfs (32B = 2 x b128) global -> LDS.
__device__ __forceinline__ void stage16(const _Float16* g, _Float16* l) {
#if ASYNC_LDS
    __builtin_amdgcn_global_load_async_to_lds_b128(
        (glob_v4i_ptr)(g), (lds_v4i_ptr)(l), 0, 0);
    __builtin_amdgcn_global_load_async_to_lds_b128(
        (glob_v4i_ptr)(g + 8), (lds_v4i_ptr)(l + 8), 0, 0);
#else
    *(v8h*)(l)     = *(const v8h*)(g);
    *(v8h*)(l + 8) = *(const v8h*)(g + 8);
#endif
}

__device__ __forceinline__ void stage_wait() {
#if ASYNC_LDS
    asm volatile("s_wait_asynccnt 0" ::: "memory");
#endif
}

// D = A(16x32 f16) * B(32x16 f16) + C(16x16 f32)
__device__ __forceinline__ v8f wmma_f16(v16h a, v16h b, v8f c) {
    return __builtin_amdgcn_wmma_f32_16x16x32_f16(
        /*neg_a=*/false, a, /*neg_b=*/false, b,
        /*c_mod=*/(short)0, c, /*reuse_a=*/false, /*reuse_b=*/false);
}

// A-fragment (16x32, 16-bit): lane m = lane&15, element i maps to
// k = i + (i & 8) + 8*(lane>=16)  (ISA 7.12.2).  Two contiguous v8h loads.
__device__ __forceinline__ v16h load_frag_a(const _Float16* base, int ld, int lane) {
    const int hi = lane >> 4, m = lane & 15;
    const _Float16* p = base + m * ld + 8 * hi;
    v8h lo = *(const v8h*)(p);
    v8h hh = *(const v8h*)(p + 16);
    return __builtin_shufflevector(lo, hh, 0,1,2,3,4,5,6,7,8,9,10,11,12,13,14,15);
}

// B-fragment (32x16, 16-bit) from a tile stored TRANSPOSED as [n][k], ld halfs:
// lane n = lane&15, element i maps to k = i + (i & 8) + 8*(lane>=16).  Two contiguous v8h loads.
__device__ __forceinline__ v16h load_frag_b(const _Float16* base, int ld, int lane) {
    const int hi = lane >> 4, n = lane & 15;
    const _Float16* p = base + n * ld + 8 * hi;
    v8h b0 = *(const v8h*)(p);
    v8h b1 = *(const v8h*)(p + 16);
    return __builtin_shufflevector(b0, b1, 0,1,2,3,4,5,6,7,8,9,10,11,12,13,14,15);
}

// ---------------------------------------------------------------------------
// f32 -> f16 conversion (4 elements / thread)
// ---------------------------------------------------------------------------
__global__ void cvt_f32_f16(const float* __restrict__ in, _Float16* __restrict__ out, int n) {
    int i = (blockIdx.x * blockDim.x + threadIdx.x) * 4;
    if (i < n) {
        float4 v = *(const float4*)(in + i);
        v4h h;
        h.x = (_Float16)v.x; h.y = (_Float16)v.y;
        h.z = (_Float16)v.z; h.w = (_Float16)v.w;
        *(v4h*)(out + i) = h;
    }
}

// ---------------------------------------------------------------------------
// Tiled GEMM: Y[M,N] = A[M,K] * W[N,K]^T   (A,W f16; Y f16 or f32)
// 256 threads = 8 waves. Block tile 128x128, K stepped by 32, double-buffered
// LDS staging via async-to-LDS. Wave (wave>>1, wave&1) computes 32x64.
// ---------------------------------------------------------------------------
template <bool F16OUT>
__global__ __launch_bounds__(256) void gemm_xwT(const _Float16* __restrict__ A,
                                                const _Float16* __restrict__ W,
                                                void* __restrict__ Cout,
                                                int M, int N, int K) {
    __shared__ alignas(16) _Float16 ldsA[2][128 * 32];
    __shared__ alignas(16) _Float16 ldsB[2][128 * 32];

    const int tid  = threadIdx.x;
    const int lane = tid & 31;
    const int wave = tid >> 5;
    const int tm0  = blockIdx.y * 128;
    const int tn0  = blockIdx.x * 128;
    const int wm0  = (wave >> 1) * 32;
    const int wn0  = (wave & 1) * 64;

    v8f acc[2][4] = {};

    const int lrow = tid >> 1;        // 0..127
    const int lcol = (tid & 1) * 16;  // 0 or 16

    auto stage_tiles = [&](int buf, int ks) {
        const _Float16* ga = A + (size_t)(tm0 + lrow) * K + ks + lcol;
        stage16(ga, &ldsA[buf][lrow * 32 + lcol]);
        const _Float16* gb = W + (size_t)(tn0 + lrow) * K + ks + lcol;
        stage16(gb, &ldsB[buf][lrow * 32 + lcol]);
    };

    // Prologue: fill buffer 0
    stage_tiles(0, 0);
    stage_wait();
    __syncthreads();

    int p = 0;
    for (int ks = 0; ks < K; ks += 32) {
        // Kick off the next tile into the other buffer while we compute.
        if (ks + 32 < K) {
            stage_tiles(p ^ 1, ks + 32);
            __builtin_prefetch(A + (size_t)(tm0 + lrow) * K + ks + 64 + lcol, 0, 1);
        }

        v16h afr0 = load_frag_a(&ldsA[p][(wm0)      * 32], 32, lane);
        v16h afr1 = load_frag_a(&ldsA[p][(wm0 + 16) * 32], 32, lane);
#pragma unroll
        for (int nt = 0; nt < 4; nt++) {
            v16h bfr = load_frag_b(&ldsB[p][(wn0 + nt * 16) * 32], 32, lane);
            acc[0][nt] = wmma_f16(afr0, bfr, acc[0][nt]);
            acc[1][nt] = wmma_f16(afr1, bfr, acc[1][nt]);
        }

        stage_wait();        // next tile fully in LDS (ASYNCcnt == 0)
        __syncthreads();     // everyone done reading buf p, next tile published
        p ^= 1;
    }

    // C/D layout: row = r + 8*(lane>=16), col = lane&15 (ISA 7.12.2)
    const int hi = lane >> 4, n = lane & 15;
#pragma unroll
    for (int mt = 0; mt < 2; mt++)
#pragma unroll
        for (int nt = 0; nt < 4; nt++) {
            const int col = tn0 + wn0 + nt * 16 + n;
#pragma unroll
            for (int r = 0; r < 8; r++) {
                const int row = tm0 + wm0 + mt * 16 + r + 8 * hi;
                if (F16OUT)
                    ((_Float16*)Cout)[(size_t)row * N + col] = (_Float16)acc[mt][nt][r];
                else
                    ((float*)Cout)[(size_t)row * N + col] = acc[mt][nt][r];
            }
        }
}

// ---------------------------------------------------------------------------
// Flash attention: block = (head, 64 q-rows), 128 threads = 4 waves.
// Each wave owns 16 q-rows; key blocks of 32 staged in LDS (K via async-to-LDS,
// V transposed through ds_store so the P*V B-fragment is contiguous).
// Online softmax in f32 with wave32 shuffle reductions.
// ---------------------------------------------------------------------------
__global__ __launch_bounds__(128) void flash_attn(const _Float16* __restrict__ Qh,
                                                  const _Float16* __restrict__ Kh,
                                                  const _Float16* __restrict__ Vh,
                                                  _Float16* __restrict__ Oh) {
    __shared__ alignas(16) _Float16 ldsK[32 * 128];      // [key][d]
    __shared__ alignas(16) _Float16 ldsVt[128 * 32];     // [d][key]  (transposed)
    __shared__ alignas(16) _Float16 ldsP[4][16 * 32];    // per-wave P (m x 32 keys)

    const int tid  = threadIdx.x;
    const int lane = tid & 31;
    const int wave = tid >> 5;
    const int h    = blockIdx.y;       // q head (kv-major: h = kv*4 + g)
    const int kvh  = h >> 2;
    const int q0   = blockIdx.x * 64 + wave * 16;
    const int hi   = lane >> 4, ln = lane & 15;

    // Q tile 16x128 held in registers as 4 A-fragments (K-chunks of 32)
    v16h qf[4];
#pragma unroll
    for (int c = 0; c < 4; c++) {
        const _Float16* p = Qh + (size_t)(q0 + ln) * HIDDEN + h * HD + c * 32 + 8 * hi;
        v8h lo = *(const v8h*)(p);
        v8h hh = *(const v8h*)(p + 16);
        qf[c] = __builtin_shufflevector(lo, hh, 0,1,2,3,4,5,6,7,8,9,10,11,12,13,14,15);
    }

    float mrow[8], lrow[8];
    v8f acc[8] = {};
#pragma unroll
    for (int r = 0; r < 8; r++) { mrow[r] = -1e30f; lrow[r] = 0.0f; }

    const float sc = 0.08838834764831845f;  // 1/sqrt(HD)

    for (int kb = 0; kb < SEQ; kb += 32) {
        // ---- stage K block [32 x 128] (async) and V^T block [128 x 32] ----
        {
            const int key = tid >> 2;          // 0..31
            const int d0  = (tid & 3) * 32;    // 0,32,64,96
            const _Float16* gk = Kh + (size_t)(kb + key) * KVDIM + kvh * HD + d0;
            stage16(gk,      &ldsK[key * 128 + d0]);
            stage16(gk + 16, &ldsK[key * 128 + d0 + 16]);
            const _Float16* gv = Vh + (size_t)(kb + key) * KVDIM + kvh * HD + d0;
#pragma unroll
            for (int j = 0; j < 32; j++)
                ldsVt[(d0 + j) * 32 + key] = gv[j];
        }
        stage_wait();
        __syncthreads();

        // ---- S = Q * K^T : two 16-key tiles, 4 K-chunks each ----
        v8f s0 = {}, s1 = {};
#pragma unroll
        for (int c = 0; c < 4; c++) {
            v16h b0 = load_frag_b(&ldsK[0  * 128 + c * 32], 128, lane);
            v16h b1 = load_frag_b(&ldsK[16 * 128 + c * 32], 128, lane);
            s0 = wmma_f16(qf[c], b0, s0);
            s1 = wmma_f16(qf[c], b1, s1);
        }

        // ---- online softmax (rows m = r + 8*hi live in one 16-lane half) ----
        float p0[8], p1[8], csc[8];
#pragma unroll
        for (int r = 0; r < 8; r++) {
            float a0 = s0[r] * sc, a1 = s1[r] * sc;
            float v = fmaxf(a0, a1);
            v = fmaxf(v, __shfl_xor(v, 1));
            v = fmaxf(v, __shfl_xor(v, 2));
            v = fmaxf(v, __shfl_xor(v, 4));
            v = fmaxf(v, __shfl_xor(v, 8));
            float mn = fmaxf(mrow[r], v);
            csc[r] = __expf(mrow[r] - mn);
            p0[r]  = __expf(a0 - mn);
            p1[r]  = __expf(a1 - mn);
            float rs = p0[r] + p1[r];
            rs += __shfl_xor(rs, 1);
            rs += __shfl_xor(rs, 2);
            rs += __shfl_xor(rs, 4);
            rs += __shfl_xor(rs, 8);
            lrow[r] = lrow[r] * csc[r] + rs;
            mrow[r] = mn;
        }
#pragma unroll
        for (int nc = 0; nc < 8; nc++)
#pragma unroll
            for (int r = 0; r < 8; r++) acc[nc][r] *= csc[r];

        // ---- C-layout -> A-layout transpose of P via per-wave LDS ----
        _Float16* P = &ldsP[wave][0];
#pragma unroll
        for (int r = 0; r < 8; r++) {
            const int m = r + 8 * hi;
            P[m * 32 + ln]      = (_Float16)p0[r];
            P[m * 32 + 16 + ln] = (_Float16)p1[r];
        }
        asm volatile("s_wait_dscnt 0" ::: "memory");  // same-wave LDS RAW
        v16h pf = load_frag_a(P, 32, lane);

        // ---- O += P * V : 8 d-chunks of 16 ----
#pragma unroll
        for (int nc = 0; nc < 8; nc++) {
            v16h vf = load_frag_b(&ldsVt[(nc * 16) * 32], 32, lane);
            acc[nc] = wmma_f16(pf, vf, acc[nc]);
        }
        __syncthreads();
    }

    // ---- epilogue: O[m][d] = acc / l, f16 out at column h*128 + d ----
#pragma unroll
    for (int nc = 0; nc < 8; nc++) {
        const int col = h * HD + nc * 16 + ln;
#pragma unroll
        for (int r = 0; r < 8; r++) {
            const int row = q0 + r + 8 * hi;
            Oh[(size_t)row * HIDDEN + col] = (_Float16)(acc[nc][r] / lrow[r]);
        }
    }
}

// ---------------------------------------------------------------------------
// Host launcher
// ---------------------------------------------------------------------------
extern "C" void kernel_launch(void* const* d_in, const int* in_sizes, int n_in,
                              void* d_out, int out_size, void* d_ws, size_t ws_size,
                              hipStream_t stream) {
    const float* x  = (const float*)d_in[0];
    const float* Wq = (const float*)d_in[1];
    const float* Wk = (const float*)d_in[2];
    const float* Wv = (const float*)d_in[3];
    const float* Wo = (const float*)d_in[4];
    float* out = (float*)d_out;

    _Float16* ws = (_Float16*)d_ws;
    size_t o = 0;
    _Float16* xh  = ws + o; o += (size_t)SEQ * HIDDEN;
    _Float16* wqh = ws + o; o += (size_t)HIDDEN * HIDDEN;
    _Float16* wkh = ws + o; o += (size_t)KVDIM * HIDDEN;
    _Float16* wvh = ws + o; o += (size_t)KVDIM * HIDDEN;
    _Float16* woh = ws + o; o += (size_t)HIDDEN * HIDDEN;
    _Float16* qh  = ws + o; o += (size_t)SEQ * HIDDEN;
    _Float16* kh  = ws + o; o += (size_t)SEQ * KVDIM;
    _Float16* vh  = ws + o; o += (size_t)SEQ * KVDIM;
    _Float16* oh  = ws + o; o += (size_t)SEQ * HIDDEN;

    auto cvt = [&](const float* src, _Float16* dst, int n) {
        cvt_f32_f16<<<(n / 4 + 255) / 256, 256, 0, stream>>>(src, dst, n);
    };
    cvt(x,  xh,  SEQ * HIDDEN);
    cvt(Wq, wqh, HIDDEN * HIDDEN);
    cvt(Wk, wkh, KVDIM * HIDDEN);
    cvt(Wv, wvh, KVDIM * HIDDEN);
    cvt(Wo, woh, HIDDEN * HIDDEN);

    // Projections: Y = X * W^T  (f16 outputs for the WMMA attention stage)
    gemm_xwT<true><<<dim3(HIDDEN / 128, SEQ / 128), 256, 0, stream>>>(xh, wqh, qh, SEQ, HIDDEN, HIDDEN);
    gemm_xwT<true><<<dim3(KVDIM  / 128, SEQ / 128), 256, 0, stream>>>(xh, wkh, kh, SEQ, KVDIM,  HIDDEN);
    gemm_xwT<true><<<dim3(KVDIM  / 128, SEQ / 128), 256, 0, stream>>>(xh, wvh, vh, SEQ, KVDIM,  HIDDEN);

    // Attention: 16 heads x 64 q-blocks of 64 rows
    flash_attn<<<dim3(SEQ / 64, QH), 128, 0, stream>>>(qh, kh, vh, oh);

    // Output projection (f32 result)
    gemm_xwT<false><<<dim3(HIDDEN / 128, SEQ / 128), 256, 0, stream>>>(oh, woh, (void*)out, SEQ, HIDDEN, HIDDEN);
}
